// ConvPixelInceptionConvDecoder_2Layer_27023934226690
// MI455X (gfx1250) — matrix operand order repaired and checked by reading, not compiled, not run on hardware
//
#include <hip/hip_runtime.h>
#include <hip/hip_bf16.h>
#include <math.h>

// ---------------------------------------------------------------------------
// Problem constants (B=1, C=256, H=8, W=32, D=32, 8 heads x 32 head-dim)
// ---------------------------------------------------------------------------
#define C_DIM  256
#define NPIX   8192      // H*W*D
#define NH     8
#define HD     32
#define H_DIM  8
#define W_DIM  32
#define D_DIM  32
#define EPSV   1e-6f
#define SCALE  0.17677669529663687f   // 32^-0.5

// ---------------------------------------------------------------------------
// WMMA types (CDNA5 / gfx1250, wave32)
// ---------------------------------------------------------------------------
typedef __attribute__((ext_vector_type(16))) __bf16 bf16x16;
typedef __attribute__((ext_vector_type(8)))  float  f32x8;

union Frag16 { bf16x16 v; uint4 q[2]; };

__device__ __forceinline__ unsigned short f2bf(float f) {
  unsigned int u = __builtin_bit_cast(unsigned int, f);
  u += 0x7fffu + ((u >> 16) & 1u);           // round-to-nearest-even
  return (unsigned short)(u >> 16);
}

// ---------------------------------------------------------------------------
// bf16 WMMA GEMM:  C[M,N] = A[M,K] @ B[K,N] (+ bias), fp32 in/out, bf16 MACs.
// Workgroup tile 128x128, 8 waves, each wave 64x32 (4x2 frags of 16x16),
// K-step 32 == one v_wmma_f32_16x16x32_bf16 per frag pair.
// ---------------------------------------------------------------------------
#define BM 128
#define BN 128
#define BK 32
#define LDP 40   // LDS pitch in bf16 elements (80B rows -> 16B-aligned chunks)

__global__ __launch_bounds__(256) void gemm_bf16_wmma(
    const float* __restrict__ A,   // M x K row-major
    const float* __restrict__ B,   // K x N row-major
    float*       __restrict__ C,   // M x N row-major
    const float* __restrict__ bias,
    int M, int N, int K)
{
  __shared__ __align__(16) unsigned short As[BM][LDP];  // [m][k] bf16 bits
  __shared__ __align__(16) unsigned short Bs[BN][LDP];  // [n][k] (transposed)

  const int tid   = threadIdx.x;
  const int lane  = tid & 31;
  const int wave  = tid >> 5;          // 0..7
  const int waveM = wave & 1;          // 2 row-groups of 64
  const int waveN = wave >> 1;         // 4 col-groups of 32
  const int half  = lane >> 4;         // ISA A/B frag half-select
  const int l16   = lane & 15;
  const int m0 = blockIdx.y * BM;
  const int n0 = blockIdx.x * BN;

  f32x8 acc[4][2] = {};

  for (int k0 = 0; k0 < K; k0 += BK) {
    // ---- stage A tile 128x32 (fp32 -> bf16), 16 elems / thread ----
    {
      const int m = tid >> 1;
      const int k = (tid & 1) * 16;
      const float* src = A + (size_t)(m0 + m) * K + k0 + k;
      if (k0 + BK < K)
        __builtin_prefetch(src + BK, 0, 1);          // global_prefetch_b8
      #pragma unroll
      for (int i = 0; i < 4; ++i) {
        float4 f = reinterpret_cast<const float4*>(src)[i];
        unsigned short* dst = &As[m][k + i * 4];
        dst[0] = f2bf(f.x); dst[1] = f2bf(f.y);
        dst[2] = f2bf(f.z); dst[3] = f2bf(f.w);
      }
    }
    // ---- stage B tile 32x128 transposed into [n][k] ----
    {
      const int k = tid >> 3;
      const int n = (tid & 7) * 16;
      const float* src = B + (size_t)(k0 + k) * N + n0 + n;
      if (k0 + BK < K)
        __builtin_prefetch(src + (size_t)BK * N, 0, 1);
      #pragma unroll
      for (int i = 0; i < 4; ++i) {
        float4 f = reinterpret_cast<const float4*>(src)[i];
        Bs[n + i * 4 + 0][k] = f2bf(f.x);
        Bs[n + i * 4 + 1][k] = f2bf(f.y);
        Bs[n + i * 4 + 2][k] = f2bf(f.z);
        Bs[n + i * 4 + 3][k] = f2bf(f.w);
      }
    }
    __syncthreads();

    // ---- B fragments: lane holds col N=l16, K chunks [8h,8h+8)+[16+8h,..) ----
    Frag16 fb[2];
    #pragma unroll
    for (int nf = 0; nf < 2; ++nf) {
      const int n = waveN * 32 + nf * 16 + l16;
      fb[nf].q[0] = *reinterpret_cast<const uint4*>(&Bs[n][8 * half]);
      fb[nf].q[1] = *reinterpret_cast<const uint4*>(&Bs[n][16 + 8 * half]);
    }
    // ---- A fragments: lane holds row M=l16, K chunks [8h,8h+8)+[16+8h,..) ----
    #pragma unroll
    for (int mf = 0; mf < 4; ++mf) {
      const int m = waveM * 64 + mf * 16 + l16;
      Frag16 fa;
      fa.q[0] = *reinterpret_cast<const uint4*>(&As[m][8 * half]);
      fa.q[1] = *reinterpret_cast<const uint4*>(&As[m][16 + 8 * half]);
      #pragma unroll
      for (int nf = 0; nf < 2; ++nf)
        acc[mf][nf] = __builtin_amdgcn_wmma_f32_16x16x32_bf16(
            false, fa.v, false, fb[nf].v, (short)0, acc[mf][nf], false, false);
    }
    __syncthreads();
  }

  // ---- epilogue: D layout M = 8*half + r, N = l16 ----
  #pragma unroll
  for (int mf = 0; mf < 4; ++mf)
    #pragma unroll
    for (int nf = 0; nf < 2; ++nf) {
      const int n = n0 + waveN * 32 + nf * 16 + l16;
      #pragma unroll
      for (int r = 0; r < 8; ++r) {
        const int m = m0 + waveM * 64 + mf * 16 + half * 8 + r;
        float v = acc[mf][nf][r];
        if (bias) v += bias[m];
        C[(size_t)m * N + n] = v;
      }
    }
}

// ---------------------------------------------------------------------------
// q softmax over the 8192 spatial tokens of one channel row (+EPS), in place.
// ---------------------------------------------------------------------------
__global__ __launch_bounds__(256) void softmax_rows(float* __restrict__ Q)
{
  __shared__ float red[256];
  float* r = Q + (size_t)blockIdx.x * NPIX;
  const int t = threadIdx.x;

  float m = -INFINITY;
  for (int i = t; i < NPIX; i += 256) m = fmaxf(m, r[i]);
  red[t] = m; __syncthreads();
  for (int s = 128; s > 0; s >>= 1) {
    if (t < s) red[t] = fmaxf(red[t], red[t + s]);
    __syncthreads();
  }
  m = red[0]; __syncthreads();

  float sum = 0.0f;
  for (int i = t; i < NPIX; i += 256) sum += __expf(r[i] - m);
  red[t] = sum; __syncthreads();
  for (int s = 128; s > 0; s >>= 1) {
    if (t < s) red[t] += red[t + s];
    __syncthreads();
  }
  const float inv = 1.0f / red[0];
  for (int i = t; i < NPIX; i += 256) r[i] = __expf(r[i] - m) * inv + EPSV;
}

// ---------------------------------------------------------------------------
// k softmax over the 32 head channels of one (head, pixel) column (+EPS).
// One thread per (head, pixel); lanes walk contiguous pixels -> coalesced.
// ---------------------------------------------------------------------------
__global__ __launch_bounds__(256) void softmax_chan(float* __restrict__ K)
{
  const int gid = blockIdx.x * 256 + threadIdx.x;   // 65536 total
  const int n = gid >> 13;
  const int p = gid & (NPIX - 1);
  float* base = K + (size_t)n * HD * NPIX + p;

  float m = -INFINITY;
  float e[HD];
  #pragma unroll
  for (int f = 0; f < HD; ++f) m = fmaxf(m, base[(size_t)f * NPIX]);
  float s = 0.0f;
  #pragma unroll
  for (int f = 0; f < HD; ++f) { e[f] = __expf(base[(size_t)f * NPIX] - m); s += e[f]; }
  const float inv = 1.0f / s;
  #pragma unroll
  for (int f = 0; f < HD; ++f) base[(size_t)f * NPIX] = e[f] * inv + EPSV;
}

// ---------------------------------------------------------------------------
// Separable 3-D box sum over one H x W x D volume held as:
//   wave = h slice (8 waves), lane = d (32 lanes), regs v[j] = w line.
// D axis: wave32 lane inclusive scan + windowed difference (shuffles).
// W axis: serial in-register prefix (unrolled -> constant indices).
// H axis: 32KB LDS slab exchange across the 8 waves.
// Result left in v[].
// ---------------------------------------------------------------------------
template<int WH, int WW, int WD>
__device__ __forceinline__ void box3d_inplace(
    float v[W_DIM], float (*slab)[W_DIM][D_DIM], int h, int d)
{
  // --- D axis ---
  #pragma unroll
  for (int j = 0; j < W_DIM; ++j) {
    float s = v[j];
    #pragma unroll
    for (int off = 1; off < 32; off <<= 1) {
      float t = __shfl_up(s, off, 32);
      if (d >= off) s += t;
    }
    int lo = d - WD / 2;
    int hi = lo + WD; if (hi > D_DIM) hi = D_DIM;
    int loi = (lo > 0) ? (lo - 1) : 0;
    float shi = __shfl(s, hi - 1, 32);
    float slo = __shfl(s, loi, 32);
    v[j] = shi - ((lo > 0) ? slo : 0.0f);
  }
  // --- W axis ---
  float ps[W_DIM];
  float run = 0.0f;
  #pragma unroll
  for (int j = 0; j < W_DIM; ++j) { run += v[j]; ps[j] = run; }
  #pragma unroll
  for (int j = 0; j < W_DIM; ++j) {
    const int lo = j - WW / 2;
    int hi = lo + WW; if (hi > W_DIM) hi = W_DIM;
    const int loi = (lo > 0) ? (lo - 1) : 0;
    v[j] = ps[hi - 1] - ((lo > 0) ? ps[loi] : 0.0f);
  }
  // --- H axis (LDS slab) ---
  __syncthreads();                     // protect previous round's readers
  #pragma unroll
  for (int j = 0; j < W_DIM; ++j) slab[h][j][d] = v[j];
  __syncthreads();
  int hlo = h - WH / 2; if (hlo < 0) hlo = 0;
  int hhi = h - WH / 2 + WH; if (hhi > H_DIM) hhi = H_DIM;
  #pragma unroll
  for (int j = 0; j < W_DIM; ++j) {
    float s = 0.0f;
    for (int hh = hlo; hh < hhi; ++hh) s += slab[hh][j][d];
    v[j] = s;
  }
}

// ---------------------------------------------------------------------------
// Denominator: den[n,win,p] = sum_f q[n,f,p] * Box(k)[n,f,p] + EPS
// ---------------------------------------------------------------------------
template<int WH, int WW, int WD>
__device__ __forceinline__ void den_window(
    const float* __restrict__ Qh, const float* __restrict__ Kh,
    float (*slab)[W_DIM][D_DIM], int h, int d, float acc[W_DIM])
{
  float t[W_DIM];
  for (int f = 0; f < HD; ++f) {
    const float* kf = Kh + (size_t)f * NPIX + h * 1024 + d;
    const float* qf = Qh + (size_t)f * NPIX + h * 1024 + d;
    #pragma unroll
    for (int j = 0; j < W_DIM; ++j) t[j] = kf[j * 32];
    box3d_inplace<WH, WW, WD>(t, slab, h, d);
    #pragma unroll
    for (int j = 0; j < W_DIM; ++j) acc[j] += qf[j * 32] * t[j];
  }
}

__global__ __launch_bounds__(256) void den_kernel(
    const float* __restrict__ Q, const float* __restrict__ K,
    float* __restrict__ DEN)
{
  __shared__ float slab[H_DIM][W_DIM][D_DIM];
  const int n   = blockIdx.x >> 1;
  const int win = blockIdx.x & 1;
  const int h = threadIdx.x >> 5;
  const int d = threadIdx.x & 31;
  const float* Qh = Q + (size_t)n * HD * NPIX;
  const float* Kh = K + (size_t)n * HD * NPIX;

  float acc[W_DIM] = {};
  if (win == 0) den_window<2, 4, 4>(Qh, Kh, slab, h, d, acc);
  else          den_window<4, 8, 8>(Qh, Kh, slab, h, d, acc);

  float* dst = DEN + (size_t)(n * 2 + win) * NPIX + h * 1024 + d;
  #pragma unroll
  for (int j = 0; j < W_DIM; ++j) dst[j * 32] = acc[j] + EPSV;
}

// ---------------------------------------------------------------------------
// Numerator (fused): for fixed (head n, v-channel g), loop f:
//   kv = k_f * v_g ; box-sum kv ; accumulate q_f * box ; then * scale / den.
// kv tensor (268 MB if materialized) never touches memory; K/V/Q stream
// from L2 (24 MB total, fully resident in the 192 MB L2).
// ---------------------------------------------------------------------------
template<int WH, int WW, int WD>
__device__ __forceinline__ void num_window(
    const float* __restrict__ Qh, const float* __restrict__ Kh,
    const float vreg[W_DIM], const float* __restrict__ DENw,
    float (*slab)[W_DIM][D_DIM], int h, int d, float outa[W_DIM])
{
  float num[W_DIM] = {};
  float t[W_DIM];
  for (int f = 0; f < HD; ++f) {
    const float* kf = Kh + (size_t)f * NPIX + h * 1024 + d;
    const float* qf = Qh + (size_t)f * NPIX + h * 1024 + d;
    #pragma unroll
    for (int j = 0; j < W_DIM; ++j) t[j] = kf[j * 32] * vreg[j];
    box3d_inplace<WH, WW, WD>(t, slab, h, d);
    #pragma unroll
    for (int j = 0; j < W_DIM; ++j) num[j] += qf[j * 32] * t[j];
  }
  const float* dw = DENw + h * 1024 + d;
  #pragma unroll
  for (int j = 0; j < W_DIM; ++j)
    outa[j] += num[j] * SCALE / dw[j * 32];
}

__global__ __launch_bounds__(256) void num_kernel(
    const float* __restrict__ Q, const float* __restrict__ K,
    const float* __restrict__ V, const float* __restrict__ DEN,
    float* __restrict__ OUTP)
{
  __shared__ float slab[H_DIM][W_DIM][D_DIM];
  const int n = blockIdx.x >> 5;
  const int g = blockIdx.x & 31;
  const int h = threadIdx.x >> 5;
  const int d = threadIdx.x & 31;
  const size_t head = (size_t)n * HD * NPIX;
  const float* Qh = Q + head;
  const float* Kh = K + head;
  const float* Vg = V + head + (size_t)g * NPIX + h * 1024 + d;

  float vreg[W_DIM];
  #pragma unroll
  for (int j = 0; j < W_DIM; ++j) vreg[j] = Vg[j * 32];

  float outa[W_DIM] = {};
  num_window<2, 4, 4>(Qh, Kh, vreg, DEN + (size_t)(n * 2 + 0) * NPIX, slab, h, d, outa);
  num_window<4, 8, 8>(Qh, Kh, vreg, DEN + (size_t)(n * 2 + 1) * NPIX, slab, h, d, outa);

  float* dst = OUTP + head + (size_t)g * NPIX + h * 1024 + d;
  #pragma unroll
  for (int j = 0; j < W_DIM; ++j) dst[j * 32] = outa[j];
}

// ---------------------------------------------------------------------------
// Host-side pipeline
// ---------------------------------------------------------------------------
extern "C" void kernel_launch(void* const* d_in, const int* in_sizes, int n_in,
                              void* d_out, int out_size, void* d_ws, size_t ws_size,
                              hipStream_t stream) {
  const float* x  = (const float*)d_in[0];   // (1,256,8,32,32)
  const float* Wq = (const float*)d_in[1];   // (256,256)
  const float* Wk = (const float*)d_in[2];
  const float* Wv = (const float*)d_in[3];
  const float* Wp = (const float*)d_in[4];
  const float* bp = (const float*)d_in[5];   // (256,)
  float* out = (float*)d_out;                // (1,256,8,32,32)

  // workspace carve-up (fp32): Q,K,V (8MB ea) + den (0.5MB) + pre-proj (8MB)
  float* Q    = (float*)d_ws;
  float* K    = Q    + (size_t)C_DIM * NPIX;
  float* V    = K    + (size_t)C_DIM * NPIX;
  float* DEN  = V    + (size_t)C_DIM * NPIX;
  float* OUTP = DEN  + (size_t)NH * 2 * NPIX;

  const dim3 gemm_grid(NPIX / BN, C_DIM / BM);   // 64 x 2

  // 1x1x1 conv projections (bf16 WMMA GEMMs)
  gemm_bf16_wmma<<<gemm_grid, 256, 0, stream>>>(Wq, x, Q, nullptr, C_DIM, NPIX, C_DIM);
  gemm_bf16_wmma<<<gemm_grid, 256, 0, stream>>>(Wk, x, K, nullptr, C_DIM, NPIX, C_DIM);
  gemm_bf16_wmma<<<gemm_grid, 256, 0, stream>>>(Wv, x, V, nullptr, C_DIM, NPIX, C_DIM);

  // softmaxes (+EPS folded in)
  softmax_rows<<<C_DIM, 256, 0, stream>>>(Q);
  softmax_chan<<<NH * NPIX / 256, 256, 0, stream>>>(K);

  // denominator per (head, window)
  den_kernel<<<NH * 2, 256, 0, stream>>>(Q, K, DEN);

  // fused kv-outer-product + 3D box sum + q-contraction + normalization
  num_kernel<<<NH * HD, 256, 0, stream>>>(Q, K, V, DEN, OUTP);

  // output projection with bias
  gemm_bf16_wmma<<<gemm_grid, 256, 0, stream>>>(Wp, OUTP, out, bp, C_DIM, NPIX, C_DIM);
}
